// ParallelRetention_fast3_46643344835366
// MI455X (gfx1250) — matrix operand order repaired and checked by reading, not compiled, not run on hardware
//
#include <hip/hip_runtime.h>
#include <stdint.h>

#define BB 2
#define HH 16
#define SS 2048
#define DD 128

typedef _Float16 half8  __attribute__((ext_vector_type(8)));
typedef _Float16 half16 __attribute__((ext_vector_type(16)));
typedef float    f32x8  __attribute__((ext_vector_type(8)));

union HV16 { half16 v; half8 h[2]; };

// Build a v16h WMMA fragment from two 16-byte chunks (works for LDS or global ptrs).
__device__ __forceinline__ half16 load_frag(const _Float16* p0, const _Float16* p1) {
  HV16 u;
  u.h[0] = *(const half8*)p0;
  u.h[1] = *(const half8*)p1;
  return u.v;
}

// ln(gamma_h) with gamma = 1 - 2^-(5+h), accurate near 1 via series.
__device__ __forceinline__ float ln_gamma(int h) {
  float x  = exp2f(-(float)(5 + h));       // exact power of two
  float x2 = x * x;
  return -(x + 0.5f*x2 + (1.0f/3.0f)*x2*x + 0.25f*x2*x2 + 0.2f*x2*x2*x);
}

// ---- CDNA5 async global->LDS copy (ASYNCcnt path) ----
__device__ __forceinline__ void async16(uint32_t lds_off, const void* gptr) {
  asm volatile("global_load_async_to_lds_b128 %0, %1, off"
               :: "v"(lds_off), "v"(gptr) : "memory");
}
__device__ __forceinline__ void wait_async0() {
  asm volatile("s_wait_asynccnt 0x0" ::: "memory");
}

// ---------------- pre-pass: q/L and k -> f16 ----------------
__global__ void prep_qk(const float* __restrict__ q, const float* __restrict__ k,
                        _Float16* __restrict__ qh, _Float16* __restrict__ kh) {
  size_t idx = (size_t)blockIdx.x * blockDim.x + threadIdx.x;
  size_t t   = idx >> 7;            // / D
  int i      = (int)(t & (SS - 1)); // seq position
  size_t bh  = t >> 11;             // / S
  int h      = (int)(bh & (HH - 1));
  float x    = exp2f(-(float)(5 + h));        // 1 - gamma
  float lng  = ln_gamma(h);
  float L    = -expm1f((float)(i + 1) * lng) / x;  // sum_{j<=i} gamma^(i-j)
  qh[idx] = (_Float16)(q[idx] / L);
  kh[idx] = (_Float16)k[idx];
}

// ---------------- pre-pass: v -> f16 transposed [b,h,d,j] ----------------
__global__ void prep_vt(const float* __restrict__ v, _Float16* __restrict__ vt) {
  __shared__ _Float16 tile[64][72];
  const int j0 = blockIdx.x * 64;
  const int d0 = blockIdx.y * 64;
  const size_t base = (size_t)blockIdx.z * SS * DD;
  const int tx = threadIdx.x & 63;
  const int ty = threadIdx.x >> 6;     // 0..3
#pragma unroll
  for (int rr = 0; rr < 64; rr += 4)
    tile[rr + ty][tx] = (_Float16)v[base + (size_t)(j0 + rr + ty) * DD + d0 + tx];
  __syncthreads();
#pragma unroll
  for (int rr = 0; rr < 64; rr += 4)
    vt[base + (size_t)(d0 + rr + ty) * SS + j0 + tx] = tile[tx][rr + ty];
}

// ---------------- main retention kernel ----------------
// block = 128 query rows of one (b,h); 8 waves, 16 rows per wave.
__global__ __launch_bounds__(256, 1)
void retention_main(const _Float16* __restrict__ qh, const _Float16* __restrict__ kh,
                    const _Float16* __restrict__ vt, float* __restrict__ out) {
  __shared__ _Float16 ktile[2][64 * 136];   // K tile, row-major [j][d], stride 136
  __shared__ _Float16 vtile[2][128 * 72];   // V^T tile [d][j], stride 72
  __shared__ _Float16 stile[8][16 * 72];    // per-wave score tile [m][j], stride 72

  const int b  = blockIdx.z, h = blockIdx.y;
  const int i0 = blockIdx.x * 128;
  const int tid = threadIdx.x;
  const int w   = tid >> 5;
  const int l   = tid & 31;
  const int l16 = l & 15;
  const int hi  = l >> 4;                   // lane half (0/1)
  const int imin = i0 + 16 * w;
  const int imax = imin + 15;

  const size_t head = ((size_t)(b * HH + h)) * SS * DD;
  const _Float16* qhead  = qh + head;
  const _Float16* khead  = kh + head;
  const _Float16* vthead = vt + head;
  float* ohead = out + head;

  // Q A-fragments (rows imin..imin+15, K=D), ISA 16-bit A layout:
  // lane<16: K chunks {k0..k0+7, k0+16..k0+23}; lane>=16: +8.
  half16 qa[4];
#pragma unroll
  for (int ks = 0; ks < 4; ++ks) {
    const _Float16* p = qhead + (size_t)(imin + l16) * DD + ks * 32 + hi * 8;
    qa[ks] = load_frag(p, p + 16);
  }

  const float lg2    = ln_gamma(h) * 1.4426950408889634f;  // log2(gamma)
  const float ginv64 = exp2f(-64.0f * lg2);                // gamma^-64
  // Separable decay: gamma^(i-j) = rowfac[r] * colfac[jts]
  //   rowfac[r]  = gamma^(i)      (advanced by gamma^-64 each j-tile)
  //   colfac[jts]= gamma^-(j-j0)  (loop-invariant per lane)
  float rowfac[8];
#pragma unroll
  for (int r = 0; r < 8; ++r)
    rowfac[r] = exp2f(lg2 * (float)(imin + r + 8 * hi));
  float colfac[4];
#pragma unroll
  for (int jts = 0; jts < 4; ++jts)
    colfac[jts] = exp2f(-lg2 * (float)(16 * jts + l16));

  f32x8 oacc[8] = {};
  float rs[8] = {0.f, 0.f, 0.f, 0.f, 0.f, 0.f, 0.f, 0.f};

  const uint32_t kb[2] = {(uint32_t)(uintptr_t)&ktile[0][0], (uint32_t)(uintptr_t)&ktile[1][0]};
  const uint32_t vb[2] = {(uint32_t)(uintptr_t)&vtile[0][0], (uint32_t)(uintptr_t)&vtile[1][0]};

  // stage one 64-key tile (K: 64x128 f16 = 1024 x16B, V^T: 128x64 f16 = 1024 x16B)
  auto stage = [&](int buf, int j0) {
#pragma unroll
    for (int i = 0; i < 4; ++i) {
      int c = tid + 256 * i;
      int row = c >> 4, c16 = c & 15;
      async16(kb[buf] + row * 272 + c16 * 16,
              khead + (size_t)(j0 + row) * DD + c16 * 8);
    }
#pragma unroll
    for (int i = 0; i < 4; ++i) {
      int c = tid + 256 * i;
      int row = c >> 3, c8 = c & 7;
      async16(vb[buf] + row * 144 + c8 * 16,
              vthead + (size_t)row * SS + j0 + c8 * 8);
    }
  };

  const int jt_top = i0 / 64 + 1;
  stage(0, 0);

  for (int jt = 0; jt <= jt_top; ++jt) {
    wait_async0();
    __syncthreads();
    if (jt < jt_top) stage((jt + 1) & 1, (jt + 1) * 64);

    const int j0 = jt * 64;
    if (j0 <= imax) {
      const _Float16* kt  = &ktile[jt & 1][0];
      const _Float16* vtt = &vtile[jt & 1][0];
      _Float16* sw = &stile[w][0];
      const bool partial = (j0 + 63 > imin);

      // ---- scores: S = (Q/L) @ K^T ----
      // ks outer / jts inner: 4 independent WMMA chains -> no D->C hazard NOPs.
      f32x8 c[4] = {};
#pragma unroll
      for (int ks = 0; ks < 4; ++ks) {
#pragma unroll
        for (int jts = 0; jts < 4; ++jts) {
          const _Float16* bp = kt + (16 * jts + l16) * 136 + ks * 32 + hi * 8;
          half16 bf = load_frag(bp, bp + 16);
          c[jts] = __builtin_amdgcn_wmma_f32_16x16x32_f16(false, qa[ks], false, bf,
                                                          (short)0, c[jts], false, false);
        }
      }
      // decay + causal mask + rowsum + pack to LDS (f16)
#pragma unroll
      for (int jts = 0; jts < 4; ++jts) {
        const int j = j0 + 16 * jts + l16;
        const float cf = colfac[jts];
#pragma unroll
        for (int r = 0; r < 8; ++r) {
          float s = c[jts][r] * (rowfac[r] * cf);
          if (partial && (j > imin + r + 8 * hi)) s = 0.0f;
          rs[r] += s;
          sw[(r + 8 * hi) * 72 + 16 * jts + l16] = (_Float16)s;
        }
      }
      // decay advance: gamma^i -> gamma^(i-64) relative to next j-tile
#pragma unroll
      for (int r = 0; r < 8; ++r) rowfac[r] *= ginv64;

      // ---- O += S @ V (8 independent WMMA chains) ----
#pragma unroll
      for (int ks = 0; ks < 2; ++ks) {
        const _Float16* ap = sw + l16 * 72 + ks * 32 + hi * 8;
        half16 af = load_frag(ap, ap + 16);
#pragma unroll
        for (int dt = 0; dt < 8; ++dt) {
          const _Float16* bp = vtt + (dt * 16 + l16) * 72 + ks * 32 + hi * 8;
          half16 bf = load_frag(bp, bp + 16);
          oacc[dt] = __builtin_amdgcn_wmma_f32_16x16x32_f16(false, af, false, bf,
                                                            (short)0, oacc[dt], false, false);
        }
      }
    }
  }

  // row-sum reduce across the 16 lanes of each half (rows match C layout)
#pragma unroll
  for (int r = 0; r < 8; ++r) {
    float vv = rs[r];
    vv += __shfl_xor(vv, 1, 32);
    vv += __shfl_xor(vv, 2, 32);
    vv += __shfl_xor(vv, 4, 32);
    vv += __shfl_xor(vv, 8, 32);
    rs[r] = vv;
  }
  float pinv[8];
#pragma unroll
  for (int r = 0; r < 8; ++r) pinv[r] = 1.0f / fmaxf(1.0f, fabsf(rs[r]));

#pragma unroll
  for (int dt = 0; dt < 8; ++dt)
#pragma unroll
    for (int r = 0; r < 8; ++r) {
      int i = imin + r + 8 * hi;
      ohead[(size_t)i * DD + dt * 16 + l16] = oacc[dt][r] * pinv[r];
    }
}

extern "C" void kernel_launch(void* const* d_in, const int* in_sizes, int n_in,
                              void* d_out, int out_size, void* d_ws, size_t ws_size,
                              hipStream_t stream) {
  (void)in_sizes; (void)n_in; (void)out_size; (void)ws_size;
  const float* q = (const float*)d_in[0];
  const float* k = (const float*)d_in[1];
  const float* v = (const float*)d_in[2];
  // d_in[3] (omask) unused: decay mask and L recomputed analytically in-kernel.

  const size_t NE = (size_t)BB * HH * SS * DD;
  _Float16* qh  = (_Float16*)d_ws;
  _Float16* kh  = qh + NE;
  _Float16* vth = kh + NE;

  prep_qk<<<dim3((unsigned)(NE / 256)), 256, 0, stream>>>(q, k, qh, kh);
  prep_vt<<<dim3(SS / 64, DD / 64, BB * HH), 256, 0, stream>>>(v, vth);
  retention_main<<<dim3(SS / 128, HH, BB), 256, 0, stream>>>(qh, kh, vth, (float*)d_out);
}
